// ModulatedConv2d_20882130993775
// MI455X (gfx1250) — compile-verified, matrix operand order checked by reading, not run on hardware
//
#include <hip/hip_runtime.h>
#include <hip/hip_bf16.h>

// ModulatedConv2d (StyleGAN2) for MI455X / gfx1250.
// Compute-bound (~500 FLOP/byte) -> implicit-GEMM conv on v_wmma_f32_16x16x32_f16.
// v3: A-tile staging via GLOBAL_LOAD_ASYNC_TO_LDS_B128 with a 2-deep ping-pong
//     pipeline over the 9 taps (ASYNCcnt-throttled), overlapping weight DMA
//     with WMMA compute. Fixes the half-tile A-staging bug of v1/v2.

typedef __attribute__((ext_vector_type(16))) _Float16 v16h;
typedef __attribute__((ext_vector_type(8)))  float    v8f;

union Frag {
    v16h v;
    unsigned int d[8];
};

#define IC   256
#define OC   256
#define HH   64
#define WW   64
#define NTAP 9
#define ROWS 4            // output rows per block

// Async copy of one 16-byte chunk: global -> LDS, tracked by ASYNCcnt.
__device__ __forceinline__ void async_copy_b128(unsigned lds_off, const void* gptr) {
    unsigned long long ga = (unsigned long long)(size_t)gptr;
    asm volatile("global_load_async_to_lds_b128 %0, %1, off"
                 :: "v"(lds_off), "v"(ga)
                 : "memory");
}

// ---------------------------------------------------------------------------
// Kernel 0: s[b][ic] = dot(style[b], mod_w[ic]) / sqrt(512) + mod_b[ic]
// ---------------------------------------------------------------------------
__global__ __launch_bounds__(256)
void style_mod_kernel(const float* __restrict__ style,   // (8,512)
                      const float* __restrict__ mod_w,   // (256,512)
                      const float* __restrict__ mod_b,   // (256)
                      float* __restrict__ s)             // (8,256)
{
    const int b  = blockIdx.x;    // 0..7
    const int ic = threadIdx.x;   // 0..255
    const float lin_scale = 0.04419417382415922f;  // 1/sqrt(512)
    const float* st = style + b * 512;
    const float* mw = mod_w + ic * 512;
    float acc = 0.f;
    for (int d = 0; d < 512; ++d) acc += st[d] * mw[d];
    s[b * IC + ic] = acc * lin_scale + mod_b[ic];
}

// ---------------------------------------------------------------------------
// Kernel 1: per-sample modulated + demodulated weights, f16, [b][tap][oc][ic]
// ---------------------------------------------------------------------------
__global__ __launch_bounds__(256)
void modulate_kernel(const float* __restrict__ weight,   // (256,256,3,3)
                     const float* __restrict__ s,        // (8,256)
                     _Float16* __restrict__ wEff)        // (8,9,256,256)
{
    const int oc = blockIdx.x;    // 0..255
    const int b  = blockIdx.y;    // 0..7
    const int ic = threadIdx.x;   // 0..255
    const float conv_scale = 0.020833333333333332f;  // 1/sqrt(256*9) = 1/48

    const float sm = s[b * IC + ic] * conv_scale;
    const float* wp = weight + (oc * IC + ic) * NTAP;

    float w[NTAP];
    float sq = 0.f;
#pragma unroll
    for (int t = 0; t < NTAP; ++t) {
        w[t] = wp[t] * sm;
        sq += w[t] * w[t];
    }

    __shared__ float red[256];
    red[ic] = sq;
    __syncthreads();
#pragma unroll
    for (int off = 128; off > 0; off >>= 1) {
        if (ic < off) red[ic] += red[ic + off];
        __syncthreads();
    }
    const float dscale = rsqrtf(red[0] + 1e-8f);   // DEMOD

#pragma unroll
    for (int t = 0; t < NTAP; ++t) {
        wEff[(((size_t)b * NTAP + t) * OC + oc) * IC + ic] = (_Float16)(w[t] * dscale);
    }
}

// ---------------------------------------------------------------------------
// Kernel 2: implicit-GEMM conv with WMMA + async-LDS weight pipeline.
//   Per block: batch b, 128 output channels (oc0..+127), 4 output rows.
//   K loop: 8 ic-blocks of 32 x 9 taps; A tiles double-buffered via
//   GLOBAL_LOAD_ASYNC_TO_LDS_B128 so tap t+1's DMA overlaps tap t's WMMAs.
//   8 waves = 2(M) x 4(N); each wave: 64 oc x 16 px x 4 rows -> 16 accums,
//   16 WMMAs per tap, 1152 WMMAs per wave total.
// ---------------------------------------------------------------------------
__global__ __launch_bounds__(256)
void conv_wmma_kernel(const float* __restrict__ x,       // (8,256,64,64)
                      const _Float16* __restrict__ wEff, // (8,9,256,256)
                      float* __restrict__ out)           // (8,256,64,64)
{
    const int tileM = blockIdx.x;        // 0..1
    const int y0    = blockIdx.y * ROWS; // 0,4,..,60
    const int b     = blockIdx.z;        // 0..7
    const int oc0   = tileM * 128;

    const int tid  = threadIdx.x;
    const int lane = tid & 31;
    const int wave = tid >> 5;
    const int wm   = wave >> 2;     // 0..1 -> M offset 64*wm
    const int wn   = wave & 3;      // 0..3 -> N offset 16*wn (pixel x0)
    const int l15  = lane & 15;
    const int hi   = lane >> 4;

    // Ping-pong A tiles, one tap each: [buf][oc 0..127][ic 0..31],
    // row stride padded to 34 halves (17 dwords, gcd(17,64)=1 -> no conflicts).
    __shared__ _Float16 lA[2 * 128 * 34];       // 17408 B
    // X tile: [xrow 0..5][col 0..65][ic 0..31], ic-contiguous (pair = 1 dword),
    // col stride padded to 34. xrow r <-> global row y0 + r - 1; halo cols 0,65.
    __shared__ _Float16 lX[6 * 66 * 34];        // 26928 B

    // Issue one tap's A tile (8 KB = 512 x 16B chunks; 2 chunks per thread)
    // as async global->LDS copies into buffer bi.
    auto issueA = [&](int icbL, int tap, int bi) {
#pragma unroll
        for (int i = 0; i < 2; ++i) {
            int u = i * 256 + tid;      // 0..511
            int r = u >> 2;             // oc row 0..127
            int q = u & 3;              // 16B chunk within 64B row
            const _Float16* g = wEff +
                ((((size_t)b * NTAP + tap) * OC + (oc0 + r)) * IC
                 + icbL * 32 + q * 8);
            unsigned lds = (unsigned)(size_t)&lA[bi * (128 * 34) + r * 34 + q * 8];
            async_copy_b128(lds, g);
        }
    };

    v8f acc[ROWS * 4];
#pragma unroll
    for (int i = 0; i < ROWS * 4; ++i)
        acc[i] = (v8f){0.f,0.f,0.f,0.f,0.f,0.f,0.f,0.f};

    for (int icb = 0; icb < 8; ++icb) {
        __syncthreads();   // all reads of previous icb's lX/lA done

        // ---- prologue: start tap 0's A-tile DMA for this ic-block
        issueA(icb, 0, 0);

        // ---- stage X: 32 ic x 6 rows x 64 cols (f32 -> f16), overlaps DMA
#pragma unroll
        for (int i = 0; i < 48; ++i) {
            int flat = i * 256 + tid;        // 0..12287
            int c    = flat & 63;
            int rp   = flat >> 6;            // 0..191
            int xr   = rp % 6;
            int ic   = rp / 6;
            int yy   = y0 + xr - 1;
            float v  = 0.f;
            if (yy >= 0 && yy < HH)
                v = x[((((size_t)b * IC) + icb * 32 + ic) * HH + yy) * WW + c];
            lX[(xr * 66 + (c + 1)) * 34 + ic] = (_Float16)v;
        }
#pragma unroll
        for (int j = 0; j < 2; ++j) {        // zero halo columns (6*2*32 = 384)
            int idx = j * 256 + tid;
            if (idx < 384) {
                int xr  = idx / 64;
                int col = ((idx >> 5) & 1) ? 65 : 0;
                int ic  = idx & 31;
                lX[(xr * 66 + col) * 34 + ic] = (_Float16)0.f;
            }
        }

#pragma unroll
        for (int t = 0; t < NTAP; ++t) {
            // (a) previous compute done: buffer (t+1)&1 free to overwrite;
            //     on t==0 also makes the lX stores visible.
            __syncthreads();

            if (t < NTAP - 1) {
                issueA(icb, t + 1, (t + 1) & 1);   // next tap's DMA in flight
                // retire tap t's 2 async copies (in-order completion),
                // leave tap t+1's 2 outstanding.
                asm volatile("s_wait_asynccnt 0x2" ::: "memory");
            } else {
                asm volatile("s_wait_asynccnt 0x0" ::: "memory");
            }
            // (b) every wave has retired its tap-t copies -> tile complete.
            __syncthreads();

            const int kg = t / 3;                // ky
            const int kx = t % 3;                // col shift (x + dx + 1)
            const _Float16* abase = &lA[(t & 1) * (128 * 34)];

            // 4 A fragments, held across the 4 rows
            Frag fa[4];
#pragma unroll
            for (int mi = 0; mi < 4; ++mi) {
                const _Float16* rowp = abase + (wm * 64 + mi * 16 + l15) * 34;
#pragma unroll
                for (int p = 0; p < 8; ++p) {
                    int k = 2 * p + 8 * hi + ((p >= 4) ? 8 : 0);
                    fa[mi].d[p] = *(const unsigned int*)(rowp + k);
                }
            }
#pragma unroll
            for (int rw = 0; rw < ROWS; ++rw) {
                // input row for (output y0+rw, ky=kg): xrow = rw + kg
                Frag fb;
                const int col = wn * 16 + l15 + kx;          // 0..65
                const _Float16* colp = &lX[((rw + kg) * 66 + col) * 34];
#pragma unroll
                for (int p = 0; p < 8; ++p) {
                    int k = 2 * p + 8 * hi + ((p >= 4) ? 8 : 0);
                    fb.d[p] = *(const unsigned int*)(colp + k);
                }
#pragma unroll
                for (int mi = 0; mi < 4; ++mi) {
                    acc[rw * 4 + mi] = __builtin_amdgcn_wmma_f32_16x16x32_f16(
                        /*neg_a=*/false, fa[mi].v, /*neg_b=*/false, fb.v,
                        /*c_mod=*/(short)0, acc[rw * 4 + mi],
                        /*reuse_a=*/false, /*reuse_b=*/false);
                }
            }
        }
    }

    // ---- epilogue: C/D layout -> VGPR r: M = r + 8*hi, N = l15
    {
        const int xo = wn * 16 + l15;
#pragma unroll
        for (int rw = 0; rw < ROWS; ++rw) {
            const int yy = y0 + rw;
#pragma unroll
            for (int mi = 0; mi < 4; ++mi) {
#pragma unroll
                for (int r = 0; r < 8; ++r) {
                    int oc = oc0 + wm * 64 + mi * 16 + hi * 8 + r;
                    out[(((size_t)b * OC + oc) * HH + yy) * WW + xo] = acc[rw * 4 + mi][r];
                }
            }
        }
    }
}

// ---------------------------------------------------------------------------
extern "C" void kernel_launch(void* const* d_in, const int* in_sizes, int n_in,
                              void* d_out, int out_size, void* d_ws, size_t ws_size,
                              hipStream_t stream) {
    const float* x      = (const float*)d_in[0];   // (8,256,64,64)
    const float* style  = (const float*)d_in[1];   // (8,512)
    const float* weight = (const float*)d_in[2];   // (1,256,256,3,3)
    const float* mod_w  = (const float*)d_in[3];   // (256,512)
    const float* mod_b  = (const float*)d_in[4];   // (256,)
    float* out = (float*)d_out;

    // workspace: s (8*256 f32 = 8 KB) | wEff (8*9*256*256 f16 = 9.44 MB)
    float*    s    = (float*)d_ws;
    _Float16* wEff = (_Float16*)((char*)d_ws + 8192);

    style_mod_kernel<<<dim3(8), dim3(256), 0, stream>>>(style, mod_w, mod_b, s);
    modulate_kernel<<<dim3(256, 8), dim3(256), 0, stream>>>(weight, s, wEff);
    conv_wmma_kernel<<<dim3(2, 16, 8), dim3(256), 0, stream>>>(x, wEff, out);
}
